// SpikingQHead_1056561955254
// MI455X (gfx1250) — matrix operand order repaired and checked by reading, not compiled, not run on hardware
//
#include <hip/hip_runtime.h>

// ---------------------------------------------------------------------------
// SpikingQHead for MI455X (gfx1250, wave32).
//   k_cvt    : one-time f32 -> bf16 weight conversion into ws (Wenc, W1).
//   k_encode : fused  relu(X@Wenc^T + benc) @ W1^T + b1  -> i1 (transposed, ws)
//              bf16 WMMA (v_wmma_f32_16x16x32_bf16), f32 accumulate.
//              Block tile 64x256; all 8 B frags issued before the 8 WMMAs
//              (sched_barrier) so waits are staggered, not wait-to-zero.
//   k_lif    : 10-step LIF dynamics + tiny N=18 GEMM per step (VALU).
// ws layout: [0, 8MB)   i1T   f32 [128][16384]
//            [8MB, +2MB) WencB bf16 [256][4096]
//            [10MB,+64K) W1B   bf16 [128][256]
// ---------------------------------------------------------------------------

typedef __attribute__((ext_vector_type(16))) __bf16 v16bf;
typedef __attribute__((ext_vector_type(8)))  __bf16 v8bf;
typedef __attribute__((ext_vector_type(4)))  __bf16 v4bf;
typedef __attribute__((ext_vector_type(8)))  float  v8f;
typedef __attribute__((ext_vector_type(4)))  float  v4f;

#define HD    4096   // encoder K
#define NE    256    // encoder out / GEMM2 K
#define NH    128    // W1 rows (LIF layer 1)
#define NA    18     // actions
#define BATCH 16384
#define TS    10
#define MTILE 64     // rows per block in k_encode

#define WS_I1T_OFF   0u
#define WS_WENC_OFF  (8u * 1024u * 1024u)
#define WS_W1_OFF    (10u * 1024u * 1024u)

#if __has_builtin(__builtin_amdgcn_sched_barrier)
#define SCHED_FENCE() __builtin_amdgcn_sched_barrier(0)
#else
#define SCHED_FENCE()
#endif

static __device__ __forceinline__ v8f wmma_bf16(v16bf a, v16bf b, v8f c) {
  return __builtin_amdgcn_wmma_f32_16x16x32_bf16(
      /*neg_a=*/false, a, /*neg_b=*/false, b,
      /*c_mod=*/(short)0, c, /*reuse_a=*/false, /*reuse_b=*/false);
}

// A fragment 16x32 (MxK) from row-major f32: lane&15 = M row; lane>=16 owns
// K+8..15 / K+24..31 halves. Non-temporal b128 loads (X is streamed once) +
// packed cvt (v_cvt_pk_bf16_f32).
static __device__ __forceinline__ v16bf load_frag_a_g(const float* __restrict__ base,
                                                      int ld, int lane) {
  const int r = lane & 15, half = lane >> 4;
  const float* p0 = base + r * ld + half * 8;       // K = half*8 .. +8
  const float* p1 = p0 + 16;                        // K = 16 + half*8 .. +8
  v4f x0 = __builtin_nontemporal_load((const v4f*)(p0));
  v4f x1 = __builtin_nontemporal_load((const v4f*)(p0 + 4));
  v4f x2 = __builtin_nontemporal_load((const v4f*)(p1));
  v4f x3 = __builtin_nontemporal_load((const v4f*)(p1 + 4));
  v4bf c0 = __builtin_convertvector(x0, v4bf);
  v4bf c1 = __builtin_convertvector(x1, v4bf);
  v4bf c2 = __builtin_convertvector(x2, v4bf);
  v4bf c3 = __builtin_convertvector(x3, v4bf);
  v16bf a;
#pragma unroll
  for (int i = 0; i < 4; ++i) {
    a[i] = c0[i]; a[4 + i] = c1[i]; a[8 + i] = c2[i]; a[12 + i] = c3[i];
  }
  return a;
}

// B fragment 32x16 (KxN) from pre-converted bf16 weights, row-major [N][K]:
// lane&15 = N column, lane>=16 owns K+8..15 / K+24..31. Two b128 loads, zero cvt.
static __device__ __forceinline__ v16bf load_frag_b_bf(const __bf16* __restrict__ w,
                                                       int n0, int ld, int k, int lane) {
  const int n = n0 + (lane & 15);
  const int koff = (lane >> 4) * 8;
  const __bf16* p = w + n * ld + k + koff;
  v8bf b0 = *(const v8bf*)(p);
  v8bf b1 = *(const v8bf*)(p + 16);
  v16bf b;
#pragma unroll
  for (int i = 0; i < 8; ++i) { b[i] = b0[i]; b[8 + i] = b1[i]; }
  return b;
}

// ---------------------------------------------------------------------------
// Prep: f32 -> bf16 bulk conversion (n must be a multiple of 8).
// ---------------------------------------------------------------------------
__global__ __launch_bounds__(256, 1)
void k_cvt(const float* __restrict__ src, __bf16* __restrict__ dst, int n) {
  const int i = (blockIdx.x * blockDim.x + threadIdx.x) * 8;
  if (i >= n) return;
  v4bf c0 = __builtin_convertvector(*(const v4f*)(src + i),     v4bf);
  v4bf c1 = __builtin_convertvector(*(const v4f*)(src + i + 4), v4bf);
  v8bf o;
#pragma unroll
  for (int e = 0; e < 4; ++e) { o[e] = c0[e]; o[4 + e] = c1[e]; }
  *(v8bf*)(dst + i) = o;
}

// ---------------------------------------------------------------------------
// Kernel 1: per block, a [64 x 256] tile of `current` (bf16, LDS), then
// GEMM2 from LDS: i1[64 x 128] -> ws transposed as i1T[n][row].
// 256 threads = 8 waves. Grid = 16384/64 = 256 blocks.
// ---------------------------------------------------------------------------
__global__ __launch_bounds__(256, 1)
void k_encode(const float* __restrict__ X, const __bf16* __restrict__ WencB,
              const float* __restrict__ benc, const __bf16* __restrict__ W1B,
              const float* __restrict__ b1, float* __restrict__ i1T) {
  __shared__ __bf16 cur[MTILE][NE];  // 32 KB

  const int tid  = threadIdx.x;
  const int lane = tid & 31;
  const int wave = tid >> 5;
  const int row0 = blockIdx.x * MTILE;

  // ---------------- GEMM1: current = relu(X @ Wenc^T + benc) ----------------
  // wave -> (mt = wave>>1 in 0..3) x (nt = wave&1, 128 columns = 8 tiles each)
  const int mt = wave >> 1;
  const int nt = wave & 1;
  const float* Abase = X + (size_t)(row0 + mt * 16) * HD;

  v8f acc[8];
#pragma unroll
  for (int t = 0; t < 8; ++t)
#pragma unroll
    for (int e = 0; e < 8; ++e) acc[t][e] = 0.f;

  for (int k = 0; k < HD; k += 32) {
    // Issue A loads+cvt and ALL 16 B loads (8 fragments) first ...
    v16bf a = load_frag_a_g(Abase + k, HD, lane);
    v16bf b[8];
#pragma unroll
    for (int t = 0; t < 8; ++t)
      b[t] = load_frag_b_bf(WencB, nt * 128 + t * 16, HD, k, lane);
    SCHED_FENCE();  // ... then the 8 WMMAs: waits become staggered, and the
                    // next iteration's loads may hoist among these WMMAs.
#pragma unroll
    for (int t = 0; t < 8; ++t)
      acc[t] = wmma_bf16(a, b[t], acc[t]);
  }

  // Epilogue 1: bias + relu, stash bf16 tile in LDS.
  {
    const int half = lane >> 4;
    const int nl   = lane & 15;
#pragma unroll
    for (int t = 0; t < 8; ++t) {
      const int n  = nt * 128 + t * 16 + nl;
      const float bv = benc[n];
      v8f r;
#pragma unroll
      for (int g = 0; g < 8; ++g) {
        float v = acc[t][g] + bv;
        r[g] = v > 0.f ? v : 0.f;
      }
      v8bf rb = __builtin_convertvector(r, v8bf);
#pragma unroll
      for (int g = 0; g < 8; ++g) {
        const int m = mt * 16 + half * 8 + g;  // accum layout: VGPR g -> M
        cur[m][n] = rb[g];
      }
    }
  }
  __syncthreads();

  // ---------------- GEMM2: i1 = current @ W1^T + b1 (K = 256 from LDS) ------
  // Output [64 x 128] = 4x8 tiles; wave -> (mt2 = wave>>1) x (n20=(wave&1)*64)
  const int mt2 = wave >> 1;
  const int n20 = (wave & 1) * 64;
  const int r    = lane & 15;
  const int half = lane >> 4;

  v8f acc2[4];
#pragma unroll
  for (int t = 0; t < 4; ++t)
#pragma unroll
    for (int e = 0; e < 8; ++e) acc2[t][e] = 0.f;

  for (int k = 0; k < NE; k += 32) {
    // A fragment straight from LDS (bf16, two ds_load_b128 per lane),
    // plus all 4 B fragments, before the 4 WMMAs.
    v8bf a0 = *(const v8bf*)&cur[mt2 * 16 + r][k + half * 8];
    v8bf a1 = *(const v8bf*)&cur[mt2 * 16 + r][k + 16 + half * 8];
    v16bf a;
#pragma unroll
    for (int i = 0; i < 8; ++i) { a[i] = a0[i]; a[8 + i] = a1[i]; }
    v16bf b[4];
#pragma unroll
    for (int t = 0; t < 4; ++t)
      b[t] = load_frag_b_bf(W1B, n20 + t * 16, NE, k, lane);
    SCHED_FENCE();
#pragma unroll
    for (int t = 0; t < 4; ++t)
      acc2[t] = wmma_bf16(a, b[t], acc2[t]);
  }

  // Epilogue 2: + b1, store TRANSPOSED: i1T[n * BATCH + row].
  // Across VGPR g the M index is contiguous -> two b128 stores per tile.
  {
    const int nl = lane & 15;
#pragma unroll
    for (int t = 0; t < 4; ++t) {
      const int n  = n20 + t * 16 + nl;
      const float bv = b1[n];
      const int mbase = row0 + mt2 * 16 + half * 8;
      v4f s0, s1;
#pragma unroll
      for (int g = 0; g < 4; ++g) { s0[g] = acc2[t][g] + bv; s1[g] = acc2[t][4 + g] + bv; }
      *(v4f*)&i1T[(size_t)n * BATCH + mbase]     = s0;
      *(v4f*)&i1T[(size_t)n * BATCH + mbase + 4] = s1;
    }
  }
}

// ---------------------------------------------------------------------------
// Kernel 2: LIF dynamics. One thread per batch row. i1 is constant across
// timesteps, so layer-1 spike trains are resimulated per t (cheap, keeps
// register pressure low). W2/b2 staged in LDS; i1T reads coalesced.
// ---------------------------------------------------------------------------
__global__ __launch_bounds__(256, 1)
void k_lif(const float* __restrict__ i1T, const float* __restrict__ W2,
           const float* __restrict__ b2, float* __restrict__ out) {
  __shared__ float w2s[NA * NH];
  __shared__ float b2s[NA];

  const int tid = threadIdx.x;
  for (int i = tid; i < NA * NH; i += blockDim.x) w2s[i] = W2[i];
  if (tid < NA) b2s[tid] = b2[tid];
  __syncthreads();

  const int row = blockIdx.x * blockDim.x + tid;
  const float A = 1.f - 1.f / 0.9f;   // leak coefficient (-1/9)
  const float B = 1.f / 0.9f;         // input gain

  float v2[NA], cnt[NA];
#pragma unroll
  for (int a = 0; a < NA; ++a) { v2[a] = 0.f; cnt[a] = 0.f; }

  for (int t = 0; t < TS; ++t) {
    float i2[NA];
#pragma unroll
    for (int a = 0; a < NA; ++a) i2[a] = b2s[a];

    for (int j = 0; j < NH; ++j) {
      const float c = i1T[(size_t)j * BATCH + row];  // coalesced (thread = row)
      float v = 0.f, sp = 0.f;
      for (int s = 0; s <= t; ++s) {                 // resimulate train to step t
        v  = A * v + B * c;
        sp = (v >= 1.f) ? 1.f : 0.f;
        v  = (sp > 0.f) ? 0.f : v;
      }
      if (sp > 0.f) {
#pragma unroll
        for (int a = 0; a < NA; ++a) i2[a] += w2s[a * NH + j];
      }
    }

#pragma unroll
    for (int a = 0; a < NA; ++a) {
      float v  = A * v2[a] + B * i2[a];
      float sp = (v >= 1.f) ? 1.f : 0.f;
      cnt[a] += sp;
      v2[a] = (sp > 0.f) ? 0.f : v;
    }
  }

#pragma unroll
  for (int a = 0; a < NA; ++a) out[(size_t)row * NA + a] = cnt[a] * 0.1f;
}

// ---------------------------------------------------------------------------
extern "C" void kernel_launch(void* const* d_in, const int* in_sizes, int n_in,
                              void* d_out, int out_size, void* d_ws, size_t ws_size,
                              hipStream_t stream) {
  const float* X    = (const float*)d_in[0];   // [16384, 4096]
  const float* Wenc = (const float*)d_in[1];   // [256, 4096]
  const float* benc = (const float*)d_in[2];   // [256]
  const float* W1   = (const float*)d_in[3];   // [128, 256]
  const float* b1   = (const float*)d_in[4];   // [128]
  const float* W2   = (const float*)d_in[5];   // [18, 128]
  const float* b2   = (const float*)d_in[6];   // [18]

  char* ws = (char*)d_ws;
  float*  i1T   = (float*)(ws + WS_I1T_OFF);   // [128][16384] f32 = 8 MB
  __bf16* WencB = (__bf16*)(ws + WS_WENC_OFF); // [256][4096] bf16 = 2 MB
  __bf16* W1B   = (__bf16*)(ws + WS_W1_OFF);   // [128][256]  bf16 = 64 KB

  const int nWenc = NE * HD;   // 1048576
  const int nW1   = NH * NE;   // 32768
  k_cvt<<<(nWenc / 8 + 255) / 256, 256, 0, stream>>>(Wenc, WencB, nWenc);
  k_cvt<<<(nW1   / 8 + 255) / 256, 256, 0, stream>>>(W1,   W1B,   nW1);

  k_encode<<<BATCH / MTILE, 256, 0, stream>>>(X, WencB, benc, W1B, b1, i1T);
  k_lif<<<BATCH / 256, 256, 0, stream>>>(i1T, W2, b2, (float*)d_out);
}
